// TGMultiHeadedAttention_81870666596438
// MI455X (gfx1250) — matrix operand order repaired and checked by reading, not compiled, not run on hardware
//
#include <hip/hip_runtime.h>
#include <stdint.h>

#define B_SZ 4
#define S_LEN 1024
#define D_MODEL 1024
#define NHEAD 16
#define HDIM 64
#define NREL 257
#define NRELP 272
#define JT 32

typedef __attribute__((ext_vector_type(16))) __bf16 v16bf;
typedef __attribute__((ext_vector_type(8)))  float  v8f;
typedef __attribute__((ext_vector_type(4))) unsigned u32x4;
typedef __attribute__((ext_vector_type(8))) int      i32x8;
typedef __attribute__((ext_vector_type(4))) int      i32x4;

union BF2 { unsigned u; __bf16 h[2]; };
union FU  { unsigned u; float f; };

__device__ __forceinline__ unsigned short f32_to_bf16(float f) {
  FU x; x.f = f;
  unsigned r = x.u + 0x7FFFu + ((x.u >> 16) & 1u);
  return (unsigned short)(r >> 16);
}
__device__ __forceinline__ float bf16_to_f32(unsigned short u) {
  FU x; x.u = ((unsigned)u) << 16; return x.f;
}

// A-operand fragment (16x32 bf16). lane: hf = lane>>4, m = lane&15.
// VGPR t holds K = hf*16 + 2t + {0,1}
__device__ __forceinline__ v16bf load_afrag(const unsigned short* src, int row, int ld, int hf) {
  v16bf f;
  const unsigned short* r = src + row * ld;
#pragma unroll
  for (int t = 0; t < 8; ++t) {
    int k = (hf << 4) + (t << 1);
    BF2 b; b.u = *reinterpret_cast<const unsigned*>(r + k);
    f[2 * t] = b.h[0];
    f[2 * t + 1] = b.h[1];
  }
  return f;
}

// B-operand fragment (32x16 bf16) from a row-major [N][K] source (B transposed).
// lane: n = lane&15, hf = lane>>4; VGPR t holds K = hf*16 + 2t + {0,1}
__device__ __forceinline__ v16bf load_bfrag(const unsigned short* src, int row, int ld, int hf) {
  v16bf f;
  const unsigned short* r = src + row * ld + (hf << 4);
#pragma unroll
  for (int t = 0; t < 8; ++t) {
    BF2 b; b.u = *reinterpret_cast<const unsigned*>(r + 2 * t);
    f[2 * t] = b.h[0];
    f[2 * t + 1] = b.h[1];
  }
  return f;
}

__device__ __forceinline__ v16bf load_bfrag_f32(const float* src, int row, int ld, int hf) {
  v16bf f;
  const float* r = src + row * ld + (hf << 4);
#pragma unroll
  for (int t = 0; t < 8; ++t) {
    BF2 b;
    b.u = (unsigned)f32_to_bf16(r[2 * t]) | ((unsigned)f32_to_bf16(r[2 * t + 1]) << 16);
    f[2 * t] = b.h[0];
    f[2 * t + 1] = b.h[1];
  }
  return f;
}

__device__ __forceinline__ v8f wmma_bf16(v16bf a, v16bf b, v8f c) {
  return __builtin_amdgcn_wmma_f32_16x16x32_bf16(false, a, false, b, (short)0, c, false, false);
}

// TDM: load a [rows x 64] bf16 tile (row-major, 64-elem tensor stride) into LDS
// with 4-DWORD padding after every 32 DWORDs -> LDS row stride 72 ushorts.
__device__ __forceinline__ void tdm_load_tile_64w(unsigned lds_addr, uint64_t gaddr,
                                                  int rows) {
  u32x4 g0;
  g0.x = 1u;                                    // count=1, is_restore=0, gather=0
  g0.y = lds_addr;                              // LDS byte address
  g0.z = (unsigned)(gaddr & 0xFFFFFFFFu);       // global_addr[31:0]
  g0.w = ((unsigned)(gaddr >> 32) & 0x01FFFFFFu) | 0x80000000u;  // addr[56:32] | type=2
  i32x8 g1;
  g1[0] = (1 << 16)      // data_size = 1 (2 bytes)
        | (1 << 20)      // pad_enable
        | (4 << 22)      // pad_interval: every 32 DWORDs (128B)
        | (3 << 25);     // pad_amount: 4 DWORDs (16B)
  g1[1] = 64 << 16;      // tensor_dim0 = 64
  g1[2] = S_LEN << 16;   // tensor_dim1 = 1024
  g1[3] = 64 << 16;      // tile_dim0 = 64
  g1[4] = rows;          // tile_dim1
  g1[5] = 64;            // tensor_dim0_stride = 64 elements
  g1[6] = 0;
  g1[7] = 0;
  i32x4 z4 = {0, 0, 0, 0};
  i32x8 z8 = {0, 0, 0, 0, 0, 0, 0, 0};
  __builtin_amdgcn_tensor_load_to_lds(g0, g1, z4, z4, z8, 0);
}

// Async global -> LDS 16-byte copy (GVS mode), tracked by ASYNCcnt.
__device__ __forceinline__ void async_copy_b128(unsigned lds_off, unsigned gbyte_off,
                                                const void* base) {
  asm volatile("global_load_async_to_lds_b128 %0, %1, %2"
               :: "v"(lds_off), "v"(gbyte_off), "s"(base) : "memory");
}

// ---------------------------------------------------------------------------
// C = X(f32)[4096,1024] @ W(f32)[1024,1024] + bias, output bf16 in [B,H,S,64]
// block: 256 thr (8 waves: 4 along M x 2 along N), tile 128x128, k-step 32
// ---------------------------------------------------------------------------
__global__ __launch_bounds__(256) void proj_gemm(const float* __restrict__ X,
                                                 const float* __restrict__ W,
                                                 const float* __restrict__ bias,
                                                 unsigned short* __restrict__ out) {
  __shared__ unsigned short sA[128][40];  // [m][k]
  __shared__ unsigned short sB[128][40];  // [n][k] (W tile transposed)
  const int tid = threadIdx.x, wid = tid >> 5, lane = tid & 31;
  const int hf = lane >> 4, nl = lane & 15;
  const int m0 = blockIdx.y * 128, n0 = blockIdx.x * 128;
  const int wm = (wid & 3) * 32, wn = (wid >> 2) * 64;

  v8f acc[2][4];
#pragma unroll
  for (int mi = 0; mi < 2; ++mi)
#pragma unroll
    for (int ni = 0; ni < 4; ++ni)
      acc[mi][ni] = (v8f){0.f, 0.f, 0.f, 0.f, 0.f, 0.f, 0.f, 0.f};

#pragma unroll 1
  for (int k0 = 0; k0 < D_MODEL; k0 += 32) {
    __syncthreads();
#pragma unroll
    for (int q = 0; q < 16; ++q) {
      int idx = q * 256 + tid;            // 128x32 A elements
      int r = idx >> 5, c = idx & 31;
      sA[r][c] = f32_to_bf16(X[(size_t)(m0 + r) * D_MODEL + k0 + c]);
    }
#pragma unroll
    for (int q = 0; q < 16; ++q) {
      int idx = q * 256 + tid;            // 32x128 W elements, stored transposed
      int kk = idx >> 7, nn = idx & 127;
      sB[nn][kk] = f32_to_bf16(W[(size_t)(k0 + kk) * D_MODEL + n0 + nn]);
    }
    __syncthreads();
    v16bf a[2], bb[4];
#pragma unroll
    for (int mi = 0; mi < 2; ++mi) a[mi] = load_afrag(&sA[0][0], wm + mi * 16 + nl, 40, hf);
#pragma unroll
    for (int ni = 0; ni < 4; ++ni) bb[ni] = load_bfrag(&sB[0][0], wn + ni * 16 + nl, 40, hf);
#pragma unroll
    for (int mi = 0; mi < 2; ++mi)
#pragma unroll
      for (int ni = 0; ni < 4; ++ni)
        acc[mi][ni] = wmma_bf16(a[mi], bb[ni], acc[mi][ni]);
  }

#pragma unroll
  for (int mi = 0; mi < 2; ++mi)
#pragma unroll
    for (int ni = 0; ni < 4; ++ni) {
      int col = n0 + wn + ni * 16 + nl;
      int hh = col >> 6, dk = col & 63;
      float bv = bias[col];
#pragma unroll
      for (int v = 0; v < 8; ++v) {
        int row = m0 + wm + mi * 16 + v + (hf << 3);   // b*S + i
        int bidx = row >> 10, ii = row & 1023;
        out[(((size_t)bidx * NHEAD + hh) * S_LEN + ii) * HDIM + dk] =
            f32_to_bf16(acc[mi][ni][v] + bv);
      }
    }
}

// ---------------------------------------------------------------------------
// Fused flash attention with relative-position bias.
// Grid: B*H*(S/64) blocks of 128 threads; wave w owns 16 query rows.
// K tile staged by the Tensor Data Mover; V tile transposed via VGPR path.
// ---------------------------------------------------------------------------
__global__ __launch_bounds__(128) void attn_kernel(const unsigned short* __restrict__ Q,
                                                   const unsigned short* __restrict__ Kmat,
                                                   const unsigned short* __restrict__ V,
                                                   const int* __restrict__ mask,
                                                   const float* __restrict__ rel,
                                                   unsigned short* __restrict__ ctx) {
  __shared__ unsigned short s_k[JT][72];          // K tile [j_local][k], TDM-padded rows
  __shared__ unsigned short s_vt[HDIM][40];       // V tile transposed [d][j_local]
  __shared__ unsigned short s_p[4][16][40];       // per-wave P tiles
  __shared__ unsigned short s_dots[4][16][NRELP]; // per-wave rel-dot tables (bf16)

  const int tid = threadIdx.x, wid = tid >> 5, lane = tid & 31;
  const int hf = lane >> 4, nl = lane & 15;

  const int blk = blockIdx.x;
  const int ib = blk & 15;         // 16 i-blocks of 64 rows
  const int bh = blk >> 4;         // 0..63
  const int bidx = bh >> 4, hd = bh & 15;
  const int i0 = ib * 64 + wid * 16;

  const unsigned short* Qbh = Q + (size_t)bh * S_LEN * HDIM;
  const unsigned short* Kbh = Kmat + (size_t)bh * S_LEN * HDIM;
  const unsigned short* Vbh = V + (size_t)bh * S_LEN * HDIM;
  const int* maskb = mask + (size_t)bidx * S_LEN * S_LEN;
  const unsigned sk_lds = (unsigned)(uintptr_t)&s_k[0][0];

  // Q fragments for this wave's 16 rows (K split 0..31 / 32..63), kept in VGPRs.
  const v16bf qa0 = load_afrag(Qbh + (size_t)i0 * HDIM, nl, HDIM, hf);
  const v16bf qa1 = load_afrag(Qbh + (size_t)i0 * HDIM + 32, nl, HDIM, hf);

  // dots[m][r] = q_m . rel_emb[r]  (16 x 272, padded rows give zeros)
#pragma unroll 1
  for (int rt = 0; rt < 17; ++rt) {
    int rr = rt * 16 + nl;
    bool valid = (rr < NREL);
    int rrc = valid ? rr : 0;
    v16bf b0 = load_bfrag_f32(rel, rrc, HDIM, hf);
    v16bf b1 = load_bfrag_f32(rel + 32, rrc, HDIM, hf);
    if (!valid) {
#pragma unroll
      for (int e = 0; e < 16; ++e) { b0[e] = (__bf16)0.f; b1[e] = (__bf16)0.f; }
    }
    v8f c = (v8f){0.f, 0.f, 0.f, 0.f, 0.f, 0.f, 0.f, 0.f};
    c = wmma_bf16(qa0, b0, c);
    c = wmma_bf16(qa1, b1, c);
#pragma unroll
    for (int v = 0; v < 8; ++v)
      s_dots[wid][v + (hf << 3)][rt * 16 + nl] = f32_to_bf16(c[v]);
  }
  asm volatile("s_wait_dscnt 0" ::: "memory");

  float mrow[8], lrow[8];
  v8f acc[4];
#pragma unroll
  for (int v = 0; v < 8; ++v) { mrow[v] = -1e30f; lrow[v] = 0.f; }
#pragma unroll
  for (int ot = 0; ot < 4; ++ot) acc[ot] = (v8f){0.f, 0.f, 0.f, 0.f, 0.f, 0.f, 0.f, 0.f};

#pragma unroll 1
  for (int j0 = 0; j0 < S_LEN; j0 += JT) {
    __syncthreads();
    // K tile via Tensor Data Mover (one wave issues; TENSORcnt tracks it)
    if (wid == 0) {
      tdm_load_tile_64w(sk_lds, (uint64_t)(uintptr_t)(Kbh + (size_t)j0 * HDIM), JT);
    }
    // V tile transposed via VGPR path (overlaps with TDM)
    const unsigned short* vsrc = Vbh + (size_t)j0 * HDIM;
#pragma unroll
    for (int q = 0; q < 16; ++q) {
      int idx = q * 128 + tid;                 // 2048 elements
      int r = idx >> 6, c = idx & 63;
      s_vt[c][r] = vsrc[idx];
    }
    if (wid == 0) __builtin_amdgcn_s_wait_tensorcnt(0);
    __syncthreads();

    // scores for two 16-column tiles
    float sv[2][8];
#pragma unroll
    for (int nt = 0; nt < 2; ++nt) {
      v16bf kb0 = load_bfrag(&s_k[0][0], nt * 16 + nl, 72, hf);
      v16bf kb1 = load_bfrag(&s_k[0][32], nt * 16 + nl, 72, hf);
      v8f c = (v8f){0.f, 0.f, 0.f, 0.f, 0.f, 0.f, 0.f, 0.f};
      c = wmma_bf16(qa0, kb0, c);
      c = wmma_bf16(qa1, kb1, c);
      int j = j0 + nt * 16 + nl;
#pragma unroll
      for (int v = 0; v < 8; ++v) {
        int m = v + (hf << 3);
        int i = i0 + m;
        int d = j - i;
        d = d < -128 ? -128 : (d > 128 ? 128 : d);
        float bias = bf16_to_f32(s_dots[wid][m][d + 128]);
        float s = (c[v] + bias) * 0.125f;               // 1/sqrt(64)
        if (maskb[(size_t)i * S_LEN + j] == 0) s = -1e9f;
        sv[nt][v] = s;
      }
    }

    // online softmax over the 32 new columns (row = 16 lanes of a half-wave)
#pragma unroll
    for (int v = 0; v < 8; ++v) {
      float mx = fmaxf(sv[0][v], sv[1][v]);
#pragma unroll
      for (int off = 8; off >= 1; off >>= 1) mx = fmaxf(mx, __shfl_xor(mx, off, 32));
      float mnew = fmaxf(mrow[v], mx);
      float alpha = __expf(mrow[v] - mnew);
      float p0 = __expf(sv[0][v] - mnew);
      float p1 = __expf(sv[1][v] - mnew);
      float rs = p0 + p1;
#pragma unroll
      for (int off = 8; off >= 1; off >>= 1) rs += __shfl_xor(rs, off, 32);
      lrow[v] = lrow[v] * alpha + rs;
      mrow[v] = mnew;
#pragma unroll
      for (int ot = 0; ot < 4; ++ot) acc[ot][v] *= alpha;
      int m = v + (hf << 3);
      s_p[wid][m][nl] = f32_to_bf16(p0);
      s_p[wid][m][16 + nl] = f32_to_bf16(p1);
    }
    asm volatile("s_wait_dscnt 0" ::: "memory");

    // ctx += P(16x32) @ V(32x64)
    v16bf pa = load_afrag(&s_p[wid][0][0], nl, 40, hf);
#pragma unroll
    for (int ot = 0; ot < 4; ++ot) {
      v16bf vb = load_bfrag(&s_vt[0][0], ot * 16 + nl, 40, hf);
      acc[ot] = wmma_bf16(pa, vb, acc[ot]);
    }
  }

  // normalize and store ctx (bf16, [B,S,D] with head-concatenated columns)
#pragma unroll
  for (int v = 0; v < 8; ++v) {
    float inv = 1.0f / lrow[v];
    int i = i0 + v + (hf << 3);
    size_t rowoff = ((size_t)bidx * S_LEN + i) * D_MODEL + hd * HDIM;
#pragma unroll
    for (int ot = 0; ot < 4; ++ot)
      ctx[rowoff + ot * 16 + nl] = f32_to_bf16(acc[ot][v] * inv);
  }
}

// ---------------------------------------------------------------------------
// out(f32) = ctx(bf16)[4096,1024] @ Wo(f32) + bo
// A tile staged with global_load_async_to_lds_b128 (ASYNCcnt path)
// ---------------------------------------------------------------------------
__global__ __launch_bounds__(256) void out_gemm(const unsigned short* __restrict__ A,
                                                const float* __restrict__ W,
                                                const float* __restrict__ bias,
                                                float* __restrict__ out) {
  __shared__ unsigned short sA[128][40];
  __shared__ unsigned short sB[128][40];
  const int tid = threadIdx.x, wid = tid >> 5, lane = tid & 31;
  const int hf = lane >> 4, nl = lane & 15;
  const int m0 = blockIdx.y * 128, n0 = blockIdx.x * 128;
  const int wm = (wid & 3) * 32, wn = (wid >> 2) * 64;
  const unsigned sA_lds = (unsigned)(uintptr_t)&sA[0][0];

  v8f acc[2][4];
#pragma unroll
  for (int mi = 0; mi < 2; ++mi)
#pragma unroll
    for (int ni = 0; ni < 4; ++ni)
      acc[mi][ni] = (v8f){0.f, 0.f, 0.f, 0.f, 0.f, 0.f, 0.f, 0.f};

#pragma unroll 1
  for (int k0 = 0; k0 < D_MODEL; k0 += 32) {
    __syncthreads();
    // A tile: 128 rows x 32 bf16 = 512 x 16B chunks, async copied to LDS
#pragma unroll
    for (int q = 0; q < 2; ++q) {
      int idx = q * 256 + tid;                 // 0..511
      int r = idx >> 2;                        // row in tile
      int cb = (idx & 3) * 16;                 // byte offset within 64B row chunk
      unsigned lds_off = sA_lds + (unsigned)(r * 80 + cb);
      unsigned g_off = (unsigned)((m0 + r) * (D_MODEL * 2) + k0 * 2 + cb);
      async_copy_b128(lds_off, g_off, A);
    }
#pragma unroll
    for (int q = 0; q < 16; ++q) {
      int idx = q * 256 + tid;
      int kk = idx >> 7, nn = idx & 127;
      sB[nn][kk] = f32_to_bf16(W[(size_t)(k0 + kk) * D_MODEL + n0 + nn]);
    }
    asm volatile("s_wait_asynccnt 0x0" ::: "memory");
    __syncthreads();
    v16bf a[2], bb[4];
#pragma unroll
    for (int mi = 0; mi < 2; ++mi) a[mi] = load_afrag(&sA[0][0], wm + mi * 16 + nl, 40, hf);
#pragma unroll
    for (int ni = 0; ni < 4; ++ni) bb[ni] = load_bfrag(&sB[0][0], wn + ni * 16 + nl, 40, hf);
#pragma unroll
    for (int mi = 0; mi < 2; ++mi)
#pragma unroll
      for (int ni = 0; ni < 4; ++ni)
        acc[mi][ni] = wmma_bf16(a[mi], bb[ni], acc[mi][ni]);
  }

#pragma unroll
  for (int mi = 0; mi < 2; ++mi)
#pragma unroll
    for (int ni = 0; ni < 4; ++ni) {
      int col = n0 + wn + ni * 16 + nl;
      float bv = bias[col];
#pragma unroll
      for (int v = 0; v < 8; ++v) {
        int row = m0 + wm + mi * 16 + v + (hf << 3);
        out[(size_t)row * D_MODEL + col] = acc[mi][ni][v] + bv;
      }
    }
}

extern "C" void kernel_launch(void* const* d_in, const int* in_sizes, int n_in,
                              void* d_out, int out_size, void* d_ws, size_t ws_size,
                              hipStream_t stream) {
  const float* query = (const float*)d_in[0];
  const float* key   = (const float*)d_in[1];
  const float* value = (const float*)d_in[2];
  const int*   mask  = (const int*)d_in[3];
  const float* Wq = (const float*)d_in[4];
  const float* bq = (const float*)d_in[5];
  const float* Wk = (const float*)d_in[6];
  const float* bk = (const float*)d_in[7];
  const float* Wv = (const float*)d_in[8];
  const float* bv = (const float*)d_in[9];
  const float* Wo = (const float*)d_in[10];
  const float* bo = (const float*)d_in[11];
  const float* rel = (const float*)d_in[12];

  const size_t nElem = (size_t)B_SZ * S_LEN * D_MODEL;  // 4M
  unsigned short* Qb = (unsigned short*)d_ws;
  unsigned short* Kb = Qb + nElem;
  unsigned short* Vb = Kb + nElem;
  unsigned short* Cb = Vb + nElem;

  dim3 ggrid(D_MODEL / 128, (B_SZ * S_LEN) / 128);  // 8 x 32
  proj_gemm<<<ggrid, 256, 0, stream>>>(query, Wq, bq, Qb);
  proj_gemm<<<ggrid, 256, 0, stream>>>(key, Wk, bk, Kb);
  proj_gemm<<<ggrid, 256, 0, stream>>>(value, Wv, bv, Vb);

  attn_kernel<<<B_SZ * NHEAD * (S_LEN / 64), 128, 0, stream>>>(Qb, Kb, Vb, mask, rel, Cb);

  out_gemm<<<ggrid, 256, 0, stream>>>(Cb, Wo, bo, (float*)d_out);
}
